// MLA_16758962389096
// MI455X (gfx1250) — matrix operand order repaired and checked by reading, not compiled, not run on hardware
//
#include <hip/hip_runtime.h>
#include <cstdint>
#include <cstddef>

typedef __bf16 bf16;
typedef __attribute__((ext_vector_type(16))) __bf16 v16bf;
typedef __attribute__((ext_vector_type(8)))  float  v8f;

#define BB   2
#define TT   2048
#define DD   2048
#define NH   16
#define HD   128
#define DLAT 1024

// ---------- fragment loaders ----------
// A-frag (16x32 bf16): lane row = lane&15; elem e -> K = half*16 + e (contiguous)
static __device__ inline v16bf load_a_frag(const bf16* rowp, int half) {
  v16bf a;
  const bf16* p0 = rowp + half * 16;
  const bf16* p1 = rowp + half * 16 + 8;
#pragma unroll
  for (int e = 0; e < 8; ++e) { a[e] = p0[e]; a[8 + e] = p1[e]; }
  return a;
}
// B-frag (32x16 bf16): lane col = lane&15; elem e -> K = half*16 + e (contiguous)
static __device__ inline v16bf load_b_frag(const bf16* p) {
  return *reinterpret_cast<const v16bf*>(p);
}

// ---------- elementwise converts ----------
__global__ void cvt_bf16_kernel(const float* __restrict__ in, bf16* __restrict__ out, int n) {
  int i = blockIdx.x * blockDim.x + threadIdx.x;
  if (i < n) out[i] = (bf16)in[i];
}
// in: (K,N) row-major fp32 -> out: (N,K) row-major bf16 (i.e. B^T for contiguous B-frags)
__global__ void cvtT_bf16_kernel(const float* __restrict__ in, bf16* __restrict__ out, int K, int N) {
  int i = blockIdx.x * blockDim.x + threadIdx.x;
  if (i < K * N) {
    int k = i / N, n = i - k * N;
    out[(size_t)n * K + k] = (bf16)in[i];
  }
}

// ---------- bf16 WMMA GEMM: C(MxN) = A(MxK) * B(KxN), B given as B^T (NxK) ----------
// 8 waves/block, each wave: 16 rows x 64 cols strip. M%128==0, N%64==0, K%32==0.
__global__ __launch_bounds__(256) void gemm_bf16_kernel(
    const bf16* __restrict__ A, const bf16* __restrict__ BT,
    float* __restrict__ Cf, bf16* __restrict__ Cb, int M, int N, int K) {
  int lane = threadIdx.x & 31, wave = threadIdx.x >> 5;
  int half = lane >> 4, ln = lane & 15;
  int n0 = blockIdx.x * 64;
  int m0 = blockIdx.y * 128 + wave * 16;
  (void)M;

  v8f z = {};
  v8f acc0 = z, acc1 = z, acc2 = z, acc3 = z;
  const bf16* arow = A + (size_t)(m0 + ln) * K;
  const bf16* b0p = BT + (size_t)(n0 + 0 * 16 + ln) * K + half * 16;
  const bf16* b1p = BT + (size_t)(n0 + 1 * 16 + ln) * K + half * 16;
  const bf16* b2p = BT + (size_t)(n0 + 2 * 16 + ln) * K + half * 16;
  const bf16* b3p = BT + (size_t)(n0 + 3 * 16 + ln) * K + half * 16;

  for (int k = 0; k < K; k += 32) {
    v16bf a  = load_a_frag(arow + k, half);
    v16bf b0 = load_b_frag(b0p + k);
    v16bf b1 = load_b_frag(b1p + k);
    v16bf b2 = load_b_frag(b2p + k);
    v16bf b3 = load_b_frag(b3p + k);
    acc0 = __builtin_amdgcn_wmma_f32_16x16x32_bf16(false, a, false, b0, (short)0, acc0, false, false);
    acc1 = __builtin_amdgcn_wmma_f32_16x16x32_bf16(false, a, false, b1, (short)0, acc1, false, false);
    acc2 = __builtin_amdgcn_wmma_f32_16x16x32_bf16(false, a, false, b2, (short)0, acc2, false, false);
    acc3 = __builtin_amdgcn_wmma_f32_16x16x32_bf16(false, a, false, b3, (short)0, acc3, false, false);
  }
#pragma unroll
  for (int r = 0; r < 8; ++r) {
    size_t row = (size_t)(m0 + r + 8 * half);
    size_t base = row * N + n0 + ln;
    if (Cb) {
      Cb[base]      = (bf16)acc0[r];
      Cb[base + 16] = (bf16)acc1[r];
      Cb[base + 32] = (bf16)acc2[r];
      Cb[base + 48] = (bf16)acc3[r];
    } else {
      Cf[base]      = acc0[r];
      Cf[base + 16] = acc1[r];
      Cf[base + 32] = acc2[r];
      Cf[base + 48] = acc3[r];
    }
  }
}

// ---------- RoPE on q, in place. q: (B*T, NH*HD) bf16, interleaved pairs ----------
__global__ void rope_q_kernel(bf16* __restrict__ q) {
  int idx = blockIdx.x * blockDim.x + threadIdx.x;  // B*T*NH*64
  int i = idx & 63;
  int h = (idx >> 6) & (NH - 1);
  int row = idx >> 10;
  int t = row & (TT - 1);
  float inv = __expf(-((float)(2 * i) * (1.0f / (float)HD)) * 9.210340371976184f); // ln(10000)
  float ang = (float)t * inv;
  float c = cosf(ang), s = sinf(ang);
  size_t base = (size_t)row * (NH * HD) + h * HD + 2 * i;
  float x1 = (float)q[base], x2 = (float)q[base + 1];
  q[base]     = (bf16)(x1 * c - x2 * s);
  q[base + 1] = (bf16)(x1 * s + x2 * c);
}

// ---------- split kv_up (B*T, NH*2*HD) -> roped k (B*T, NH*HD) and vT (B,NH,HD,T) ----------
__global__ void rope_split_kv_kernel(const bf16* __restrict__ kv,
                                     bf16* __restrict__ kb, bf16* __restrict__ vT) {
  int idx = blockIdx.x * blockDim.x + threadIdx.x;  // B*T*NH*64
  int i = idx & 63;
  int h = (idx >> 6) & (NH - 1);
  int row = idx >> 10;
  int t = row & (TT - 1);
  int b = row >> 11;  // T = 2048
  float inv = __expf(-((float)(2 * i) * (1.0f / (float)HD)) * 9.210340371976184f);
  float ang = (float)t * inv;
  float c = cosf(ang), s = sinf(ang);
  size_t kvb = (size_t)row * (NH * 2 * HD) + h * (2 * HD);
  float k1 = (float)kv[kvb + 2 * i], k2 = (float)kv[kvb + 2 * i + 1];
  size_t ob = (size_t)row * (NH * HD) + h * HD + 2 * i;
  kb[ob]     = (bf16)(k1 * c - k2 * s);
  kb[ob + 1] = (bf16)(k1 * s + k2 * c);
  float v1 = (float)kv[kvb + HD + 2 * i];
  float v2 = (float)kv[kvb + HD + 2 * i + 1];
  size_t vtb = ((size_t)(b * NH + h) * HD + 2 * i) * TT + t;
  vT[vtb]      = (bf16)v1;
  vT[vtb + TT] = (bf16)v2;
}

// ---------- flash attention: one wave per 16-query tile per (b,h) ----------
__global__ __launch_bounds__(128) void mla_attn_kernel(
    const bf16* __restrict__ q, const bf16* __restrict__ k,
    const bf16* __restrict__ vT, bf16* __restrict__ o) {
  __shared__ bf16 plds[4][16 * 32];
  int lane = threadIdx.x & 31, wave = threadIdx.x >> 5;
  int half = lane >> 4, ln = lane & 15;
  int tile = blockIdx.x * 4 + wave;
  int qt = tile & (TT / 16 - 1);
  int hh = (tile >> 7) & (NH - 1);
  int b  = tile >> 11;
  int q0 = qt * 16;
  const float scale = 0.08838834764831845f;  // 1/sqrt(128)

  v16bf qa[4];
#pragma unroll
  for (int s = 0; s < 4; ++s)
    qa[s] = load_a_frag(q + (size_t)(b * TT + q0 + ln) * (NH * HD) + hh * HD + s * 32, half);

  v8f zero = {};
  v8f oacc[8];
#pragma unroll
  for (int d = 0; d < 8; ++d) oacc[d] = zero;
  float mrun[8], lrun[8];
#pragma unroll
  for (int r = 0; r < 8; ++r) { mrun[r] = -3.0e38f; lrun[r] = 0.0f; }

  const bf16* kbh = k + (size_t)b * TT * (NH * HD) + hh * HD;
  const bf16* vbh = vT + (size_t)(b * NH + hh) * HD * TT;
  bf16* pw = &plds[wave][0];

  for (int k0 = 0; k0 <= q0 + 15; k0 += 32) {
    v8f s0 = zero, s1 = zero;
#pragma unroll
    for (int s = 0; s < 4; ++s) {
      v16bf bk0 = load_b_frag(kbh + (size_t)(k0 + ln) * (NH * HD) + s * 32 + half * 16);
      v16bf bk1 = load_b_frag(kbh + (size_t)(k0 + 16 + ln) * (NH * HD) + s * 32 + half * 16);
      s0 = __builtin_amdgcn_wmma_f32_16x16x32_bf16(false, qa[s], false, bk0, (short)0, s0, false, false);
      s1 = __builtin_amdgcn_wmma_f32_16x16x32_bf16(false, qa[s], false, bk1, (short)0, s1, false, false);
    }
    bool needMask = (k0 + 31 > q0);
#pragma unroll
    for (int r = 0; r < 8; ++r) {
      float a0 = s0[r] * scale, a1 = s1[r] * scale;
      if (needMask) {
        int qi = q0 + r + 8 * half;
        if (k0 + ln > qi)      a0 = -3.0e38f;
        if (k0 + 16 + ln > qi) a1 = -3.0e38f;
      }
      s0[r] = a0; s1[r] = a1;
    }
    float alpha[8];
#pragma unroll
    for (int r = 0; r < 8; ++r) {
      float t = fmaxf(s0[r], s1[r]);
      t = fmaxf(t, __shfl_xor(t, 1));
      t = fmaxf(t, __shfl_xor(t, 2));
      t = fmaxf(t, __shfl_xor(t, 4));
      t = fmaxf(t, __shfl_xor(t, 8));
      float mnew = fmaxf(mrun[r], t);
      float p0 = __expf(s0[r] - mnew);
      float p1 = __expf(s1[r] - mnew);
      s0[r] = p0; s1[r] = p1;
      float rs = p0 + p1;
      rs += __shfl_xor(rs, 1);
      rs += __shfl_xor(rs, 2);
      rs += __shfl_xor(rs, 4);
      rs += __shfl_xor(rs, 8);
      float al = __expf(mrun[r] - mnew);
      lrun[r] = lrun[r] * al + rs;
      mrun[r] = mnew;
      alpha[r] = al;
    }
#pragma unroll
    for (int d = 0; d < 8; ++d)
#pragma unroll
      for (int r = 0; r < 8; ++r) oacc[d][r] *= alpha[r];
    // P (D-layout) -> LDS -> A-layout fragment
#pragma unroll
    for (int r = 0; r < 8; ++r) {
      pw[(r + 8 * half) * 32 + ln]      = (bf16)s0[r];
      pw[(r + 8 * half) * 32 + 16 + ln] = (bf16)s1[r];
    }
    v16bf pa;
#pragma unroll
    for (int e = 0; e < 16; ++e) {
      int kk = half * 16 + e;
      pa[e] = pw[ln * 32 + kk];
    }
#pragma unroll
    for (int d = 0; d < 8; ++d) {
      v16bf bv = load_b_frag(vbh + (size_t)(d * 16 + ln) * TT + k0 + half * 16);
      oacc[d] = __builtin_amdgcn_wmma_f32_16x16x32_bf16(false, pa, false, bv, (short)0, oacc[d], false, false);
    }
  }
#pragma unroll
  for (int r = 0; r < 8; ++r) {
    float recip = 1.0f / lrun[r];
    size_t base = (size_t)(b * TT + q0 + r + 8 * half) * (NH * HD) + hh * HD + ln;
#pragma unroll
    for (int d = 0; d < 8; ++d)
      o[base + d * 16] = (bf16)(oacc[d][r] * recip);
  }
}

// ---------- host ----------
extern "C" void kernel_launch(void* const* d_in, const int* in_sizes, int n_in,
                              void* d_out, int out_size, void* d_ws, size_t ws_size,
                              hipStream_t stream) {
  (void)in_sizes; (void)n_in; (void)out_size; (void)ws_size;
  const float* x  = (const float*)d_in[0];
  const float* wq = (const float*)d_in[1];
  const float* wd = (const float*)d_in[2];
  const float* wu = (const float*)d_in[3];
  const float* wo = (const float*)d_in[4];
  float* out = (float*)d_out;

  char* ws = (char*)d_ws;
  size_t off = 0;
  bf16* xb   = (bf16*)(ws + off); off += (size_t)BB * TT * DD * 2;        // x bf16
  bf16* wqT  = (bf16*)(ws + off); off += (size_t)DD * (NH * HD) * 2;      // wq^T
  bf16* wdT  = (bf16*)(ws + off); off += (size_t)DD * DLAT * 2;           // w_down^T
  bf16* wuT  = (bf16*)(ws + off); off += (size_t)DLAT * (2 * NH * HD) * 2;// w_up^T
  bf16* woT  = (bf16*)(ws + off); off += (size_t)(NH * HD) * DD * 2;      // wo^T
  bf16* qb   = (bf16*)(ws + off); off += (size_t)BB * TT * (NH * HD) * 2; // q (roped in place)
  bf16* ckv  = (bf16*)(ws + off); off += (size_t)BB * TT * DLAT * 2;      // c_kv
  bf16* kvr  = (bf16*)(ws + off); off += (size_t)BB * TT * (2 * NH * HD) * 2; // kv_up
  bf16* kb   = (bf16*)(ws + off); off += (size_t)BB * TT * (NH * HD) * 2; // roped k
  bf16* vTb  = (bf16*)(ws + off); off += (size_t)BB * NH * HD * TT * 2;   // v transposed
  bf16* ob   = (bf16*)(ws + off); off += (size_t)BB * TT * (NH * HD) * 2; // attention out

  const int thr = 256;
  // converts
  cvt_bf16_kernel<<<(BB * TT * DD) / thr, thr, 0, stream>>>(x, xb, BB * TT * DD);
  cvtT_bf16_kernel<<<(DD * (NH * HD)) / thr, thr, 0, stream>>>(wq, wqT, DD, NH * HD);
  cvtT_bf16_kernel<<<(DD * DLAT) / thr, thr, 0, stream>>>(wd, wdT, DD, DLAT);
  cvtT_bf16_kernel<<<(DLAT * 2 * NH * HD) / thr, thr, 0, stream>>>(wu, wuT, DLAT, 2 * NH * HD);
  cvtT_bf16_kernel<<<((NH * HD) * DD) / thr, thr, 0, stream>>>(wo, woT, NH * HD, DD);

  const int M = BB * TT;  // 4096
  // q = x @ wq
  gemm_bf16_kernel<<<dim3((NH * HD) / 64, M / 128), 256, 0, stream>>>(
      xb, wqT, nullptr, qb, M, NH * HD, DD);
  // c_kv = x @ w_down
  gemm_bf16_kernel<<<dim3(DLAT / 64, M / 128), 256, 0, stream>>>(
      xb, wdT, nullptr, ckv, M, DLAT, DD);
  // kv_up = c_kv @ w_up
  gemm_bf16_kernel<<<dim3((2 * NH * HD) / 64, M / 128), 256, 0, stream>>>(
      ckv, wuT, nullptr, kvr, M, 2 * NH * HD, DLAT);

  // RoPE + split
  rope_q_kernel<<<(BB * TT * NH * 64) / thr, thr, 0, stream>>>(qb);
  rope_split_kv_kernel<<<(BB * TT * NH * 64) / thr, thr, 0, stream>>>(kvr, kb, vTb);

  // attention: B*NH*(T/16) = 4096 wave-tiles, 4 waves/block
  mla_attn_kernel<<<(BB * NH * (TT / 16)) / 4, 128, 0, stream>>>(qb, kb, vTb, ob);

  // out = attn_out @ wo (fp32 output)
  gemm_bf16_kernel<<<dim3(DD / 64, M / 128), 256, 0, stream>>>(
      ob, woT, out, nullptr, M, DD, NH * HD);
}
